// GCN_84988812853436
// MI455X (gfx1250) — matrix operand order repaired and checked by reading, not compiled, not run on hardware
//
#include <hip/hip_runtime.h>

// ---------------- problem constants (match reference) ----------------
#define NN      100000   // nodes
#define NPAD    100096   // nodes padded to multiple of 128 (BM)
#define EE      300000   // edges
#define GG      2048     // graphs
#define RDOUT   1536     // readout width (4*256 + 512)
#define EPSBN   1e-5f

// ---------------- WMMA types ----------------
typedef __attribute__((ext_vector_type(16))) __bf16          v16bf;
typedef __attribute__((ext_vector_type(8)))  float           v8f;
typedef __attribute__((ext_vector_type(8)))  unsigned short  u16x8;
typedef __attribute__((ext_vector_type(16))) unsigned short  u16x16;

static __device__ __forceinline__ unsigned short f2bf(float f) {
    unsigned int u = __builtin_bit_cast(unsigned int, f);
    u = (u + 0x7FFFu + ((u >> 16) & 1u)) >> 16;   // round-to-nearest-even
    return (unsigned short)u;
}

static __device__ __forceinline__ v16bf frag_from(const unsigned short* p_lo,
                                                  const unsigned short* p_hi) {
    u16x8 lo = *(const u16x8*)p_lo;   // ds_load_b128
    u16x8 hi = *(const u16x8*)p_hi;   // ds_load_b128
    u16x16 v;
#pragma unroll
    for (int i = 0; i < 8; ++i) { v[i] = lo[i]; v[i + 8] = hi[i]; }
    return __builtin_bit_cast(v16bf, v);
}

static __device__ __forceinline__ void stage_async_A(unsigned lds0, unsigned lds1,
                                                     const unsigned short* g0,
                                                     const unsigned short* g1) {
    asm volatile("global_load_async_to_lds_b128 %0, %1, off"
                 :: "v"(lds0), "v"((unsigned long long)(size_t)g0) : "memory");
    asm volatile("global_load_async_to_lds_b128 %0, %1, off"
                 :: "v"(lds1), "v"((unsigned long long)(size_t)g1) : "memory");
}

// ---------------- f32 -> bf16 bulk convert (n multiple of 4) ----------------
__global__ void tobf16_kernel(const float* __restrict__ X,
                              unsigned short* __restrict__ Y, long long n) {
    long long t = ((long long)blockIdx.x * blockDim.x + threadIdx.x) * 4;
    if (t >= n) return;
    float4 v = *(const float4*)(X + t);
    unsigned lo = (unsigned)f2bf(v.x) | ((unsigned)f2bf(v.y) << 16);
    unsigned hi = (unsigned)f2bf(v.z) | ((unsigned)f2bf(v.w) << 16);
    uint2 o; o.x = lo; o.y = hi;
    *(uint2*)(Y + t) = o;
}

// ---------------- WMMA GEMM: C = A(bf16) @ W(bf16) + bias -------------------
// A readable and C writable for gridDim.x*BM rows (padded node buffers).
// Double-buffered LDS with COMPILE-TIME buffer indices (K/32 always even),
// A tile via global_load_async_to_lds_b128 (ASYNCcnt), 1 barrier per K-step.
#define BM   128
#define BN   64
#define BK   32
#define LSTR 40   // LDS row stride in ushorts (80B, 16B aligned, conflict pad)

__global__ __launch_bounds__(256)
void gemm_bf16_wmma(const unsigned short* __restrict__ A,
                    const unsigned short* __restrict__ W,
                    const float* __restrict__ bias, float* __restrict__ C,
                    int K, int D) {
    __shared__ __align__(16) unsigned short sA[2][BM * LSTR];
    __shared__ __align__(16) unsigned short sB[2][BN * LSTR];

    const int tid  = threadIdx.x;
    const int lane = tid & 31;
    const int wave = tid >> 5;
    const int wm   = (wave & 3) << 5;          // wave M offset in tile (0..96)
    const int wn   = (wave >> 2) << 5;         // wave N offset in tile (0,32)
    const int rowBase = blockIdx.x * BM;
    const int colBase = blockIdx.y * BN;

    // --- per-thread A staging: two 16B chunks (128 rows x 4 chunks of 8 bf16)
    const int ar0 = tid >> 2;                  // rows 0..63
    const int ac0 = (tid & 3) << 3;            // bf16 col 0,8,16,24
    unsigned ldsA0[2], ldsA1[2];
#pragma unroll
    for (int b = 0; b < 2; ++b) {
        ldsA0[b] = (unsigned)(size_t)&sA[b][ar0 * LSTR + ac0];
        ldsA1[b] = (unsigned)(size_t)&sA[b][(ar0 + 64) * LSTR + ac0];
    }
    const unsigned short* gA0 = A + (size_t)(rowBase + ar0)      * K + ac0;
    const unsigned short* gA1 = A + (size_t)(rowBase + ar0 + 64) * K + ac0;

    // --- per-thread B staging: one u16x8 row chunk, transposed into sB[BN][32]
    const int bk  = tid >> 3;                  // k row 0..31
    const int bn8 = (tid & 7) << 3;            // col group 0,8,..,56
    const unsigned short* gW = W + (size_t)bk * D + (colBase + bn8);

    // --- fragment offsets within one LDS buffer
    const int aOff = (wm + (lane & 15)) * LSTR + ((lane >> 4) << 3); // +0/8
    const int bOff = (wn + (lane & 15)) * LSTR + ((lane >> 4) << 3); // +0/8

    v8f acc[2][2] = {};

    // ---- prologue: stage tile 0 into buffer 0
    stage_async_A(ldsA0[0], ldsA1[0], gA0, gA1);
    {
        u16x8 wv = *(const u16x8*)gW;
#pragma unroll
        for (int i = 0; i < 8; ++i) sB[0][(bn8 + i) * LSTR + bk] = wv[i];
    }
    asm volatile("s_wait_asynccnt 0x0" ::: "memory");
    __syncthreads();

    const int nsteps = K / BK;                 // always even for our shapes

    // One pipelined step: compute from buffer CURB, prefetch k=kNext into NXTB.
#define GEMM_STEP(CURB, NXTB, kNext, doPrefetch)                               \
    {                                                                          \
        u16x8 wv;                                                              \
        if (doPrefetch) {                                                      \
            stage_async_A(ldsA0[NXTB], ldsA1[NXTB], gA0 + (kNext), gA1 + (kNext)); \
            wv = *(const u16x8*)(gW + (size_t)(kNext) * D);                    \
        }                                                                      \
        v16bf afrag[2], bfrag[2];                                              \
        _Pragma("unroll")                                                      \
        for (int mi = 0; mi < 2; ++mi)                                         \
            afrag[mi] = frag_from(&sA[CURB][aOff + mi * 16 * LSTR],            \
                                  &sA[CURB][aOff + mi * 16 * LSTR + 16]);      \
        _Pragma("unroll")                                                      \
        for (int ni = 0; ni < 2; ++ni)                                         \
            bfrag[ni] = frag_from(&sB[CURB][bOff + ni * 16 * LSTR],            \
                                  &sB[CURB][bOff + ni * 16 * LSTR + 16]);      \
        _Pragma("unroll")                                                      \
        for (int mi = 0; mi < 2; ++mi)                                         \
            _Pragma("unroll")                                                  \
            for (int ni = 0; ni < 2; ++ni)                                     \
                acc[mi][ni] = __builtin_amdgcn_wmma_f32_16x16x32_bf16(         \
                    false, afrag[mi], false, bfrag[ni],                        \
                    (short)0, acc[mi][ni], false, false);                      \
        if (doPrefetch) {                                                      \
            _Pragma("unroll")                                                  \
            for (int i = 0; i < 8; ++i)                                        \
                sB[NXTB][(bn8 + i) * LSTR + bk] = wv[i];                       \
            asm volatile("s_wait_asynccnt 0x0" ::: "memory");                  \
        }                                                                      \
        __syncthreads();                                                       \
    }

    for (int it = 0; it < nsteps; it += 2) {
        GEMM_STEP(0, 1, (it + 1) * BK, true);               // even step
        GEMM_STEP(1, 0, (it + 2) * BK, (it + 2) < nsteps);  // odd step
    }
#undef GEMM_STEP

    // --- epilogue, unguarded (C padded): C layout VGPR r -> row r (+8 hi lanes)
#pragma unroll
    for (int ni = 0; ni < 2; ++ni) {
        int col = colBase + wn + ni * 16 + (lane & 15);
        float bv = bias ? bias[col] : 0.0f;
#pragma unroll
        for (int mi = 0; mi < 2; ++mi) {
            int rbase = rowBase + wm + mi * 16 + ((lane >> 4) << 3);
            float* cp = C + (size_t)rbase * D + col;
#pragma unroll
            for (int r = 0; r < 8; ++r)
                cp[(size_t)r * D] = acc[mi][ni][r] + bv;
        }
    }
}

// ---------------- graph-structure kernels ----------------
__global__ void deg_kernel(const long long* __restrict__ ei, float* __restrict__ deg) {
    int e = blockIdx.x * blockDim.x + threadIdx.x;
    if (e < EE) atomicAdd(&deg[ei[EE + e]], 1.0f);
}

__global__ void counts_kernel(const long long* __restrict__ batch, float* __restrict__ counts) {
    int n = blockIdx.x * blockDim.x + threadIdx.x;
    if (n < NN) atomicAdd(&counts[batch[n]], 1.0f);
}

__global__ void dinv_kernel(float* __restrict__ deg_dinv) {
    int n = blockIdx.x * blockDim.x + threadIdx.x;
    if (n < NN) deg_dinv[n] = rsqrtf(deg_dinv[n] + 1.0f);
}

// agg[tgt] += hlin[src] * dinv[src]*dinv[tgt]; lc = log2(D/4) (shifts, no div)
__global__ void scatter_kernel(const float* __restrict__ hlin,
                               const long long* __restrict__ ei,
                               const float* __restrict__ dinv,
                               float* __restrict__ agg, int D, int lc, long long total) {
    long long t = (long long)blockIdx.x * blockDim.x + threadIdx.x;
    if (t >= total) return;
    long long e = t >> lc;
    int c = (int)(t & ((1 << lc) - 1)) << 2;
    long long s  = ei[e];
    long long tg = ei[EE + e];
    float nrm = dinv[s] * dinv[tg];
    float4 v = *(const float4*)(hlin + (size_t)s * D + c);
    float* ap = agg + (size_t)tg * D + c;
    atomicAdd(ap + 0, v.x * nrm);
    atomicAdd(ap + 1, v.y * nrm);
    atomicAdd(ap + 2, v.z * nrm);
    atomicAdd(ap + 3, v.w * nrm);
}

// agg <- agg + dinv^2 * hlin + bias;  accumulate per-column sum/sumsq
__global__ void combine_stats_kernel(float* __restrict__ agg,
                                     const float* __restrict__ hlin,
                                     const float* __restrict__ dinv,
                                     const float* __restrict__ bias,
                                     int rows, int D,
                                     float* __restrict__ bsum, float* __restrict__ bsq) {
    int c = threadIdx.x;               // blockDim.x == D
    float bc = bias[c], s = 0.0f, q = 0.0f;
    size_t step = (size_t)gridDim.x * D;
    const float* pl = hlin + (size_t)blockIdx.x * D + c;
    float*       pa = agg  + (size_t)blockIdx.x * D + c;
    const float* pd = dinv + blockIdx.x;
    for (int r = blockIdx.x; r < rows; r += gridDim.x) {
        float di = *pd;
        float v = *pa + di * di * (*pl) + bc;
        *pa = v;
        s += v; q += v * v;
        pl += step; pa += step; pd += gridDim.x;
    }
    atomicAdd(&bsum[c], s);
    atomicAdd(&bsq[c], q);
}

// plain per-column stats (classifier hidden)
__global__ void stats_kernel(const float* __restrict__ X, int rows, int D,
                             float* __restrict__ bsum, float* __restrict__ bsq) {
    int c = threadIdx.x;
    float s = 0.0f, q = 0.0f;
    size_t step = (size_t)gridDim.x * D;
    const float* px = X + (size_t)blockIdx.x * D + c;
    for (int r = blockIdx.x; r < rows; r += gridDim.x) {
        float v = *px;
        s += v; q += v * v;
        px += step;
    }
    atomicAdd(&bsum[c], s);
    atomicAdd(&bsq[c], q);
}

__global__ void finalize_bn_kernel(const float* __restrict__ bsum, const float* __restrict__ bsq,
                                   const float* __restrict__ gamma, const float* __restrict__ beta,
                                   float* __restrict__ scale, float* __restrict__ shift,
                                   float rows, int D) {
    int c = blockIdx.x * blockDim.x + threadIdx.x;
    if (c >= D) return;
    float mean = bsum[c] / rows;
    float var  = bsq[c] / rows - mean * mean;
    float sc = gamma[c] * rsqrtf(var + EPSBN);
    scale[c] = sc;
    shift[c] = beta[c] - mean * sc;
}

// h_out(bf16) = relu(BN(hc)); pool[batch[r]] += h_out  (into z slice)
__global__ void bn_relu_pool_kernel(const float* __restrict__ hc,
                                    const float* __restrict__ scale,
                                    const float* __restrict__ shift,
                                    const long long* __restrict__ batch,
                                    unsigned short* __restrict__ hout_bf,
                                    float* __restrict__ zslice,
                                    int rows, int D) {
    int c = threadIdx.x;
    float sc = scale[c], sh = shift[c];
    size_t step = (size_t)gridDim.x * D;
    const float*     ph = hc      + (size_t)blockIdx.x * D + c;
    unsigned short*  po = hout_bf + (size_t)blockIdx.x * D + c;
    const long long* pb = batch   + blockIdx.x;
    for (int r = blockIdx.x; r < rows; r += gridDim.x) {
        float v = fmaxf((*ph) * sc + sh, 0.0f);
        *po = f2bf(v);
        atomicAdd(&zslice[(size_t)(*pb) * RDOUT + c], v);
        ph += step; po += step; pb += gridDim.x;
    }
}

__global__ void bn_relu_kernel(float* __restrict__ X,
                               const float* __restrict__ scale,
                               const float* __restrict__ shift,
                               int rows, int D) {
    int c = threadIdx.x;
    float sc = scale[c], sh = shift[c];
    size_t step = (size_t)gridDim.x * D;
    float* px = X + (size_t)blockIdx.x * D + c;
    for (int r = blockIdx.x; r < rows; r += gridDim.x) {
        *px = fmaxf((*px) * sc + sh, 0.0f);
        px += step;
    }
}

__global__ void pool_div_kernel(float* __restrict__ z, const float* __restrict__ counts) {
    int t = blockIdx.x * blockDim.x + threadIdx.x;
    if (t < GG * RDOUT) z[t] /= fmaxf(counts[t / RDOUT], 1.0f);
}

// ---------------- host launch ----------------
static inline unsigned cdiv(long long a, long long b) { return (unsigned)((a + b - 1) / b); }

extern "C" void kernel_launch(void* const* d_in, const int* in_sizes, int n_in,
                              void* d_out, int out_size, void* d_ws, size_t ws_size,
                              hipStream_t stream) {
    (void)in_sizes; (void)n_in; (void)out_size; (void)ws_size;

    const float*     x     = (const float*)d_in[0];
    const long long* ei    = (const long long*)d_in[1];
    const long long* batch = (const long long*)d_in[2];
    const float* convW[5], *convB[5], *bnG[5], *bnB[5];
    for (int i = 0; i < 5; ++i) {
        convW[i] = (const float*)d_in[3 + i];
        convB[i] = (const float*)d_in[8 + i];
        bnG[i]   = (const float*)d_in[13 + i];
        bnB[i]   = (const float*)d_in[18 + i];
    }
    const float* w1  = (const float*)d_in[23];
    const float* b1  = (const float*)d_in[24];
    const float* g1  = (const float*)d_in[25];
    const float* be1 = (const float*)d_in[26];
    const float* w2  = (const float*)d_in[27];
    const float* b2  = (const float*)d_in[28];
    float* out = (float*)d_out;

    // workspace carve
    char* p = (char*)d_ws;
    auto carve = [&](size_t bytes) -> void* {
        void* r = (void*)p; p += (bytes + 255) & ~(size_t)255; return r;
    };
    float*          lin   = (float*)carve((size_t)NPAD * 512 * 4);   // padded rows
    float*          agg   = (float*)carve((size_t)NN * 512 * 4);
    unsigned short* xbf   = (unsigned short*)carve((size_t)NPAD * 128 * 2);
    unsigned short* hbf[2];
    hbf[0] = (unsigned short*)carve((size_t)NPAD * 512 * 2);
    hbf[1] = (unsigned short*)carve((size_t)NPAD * 512 * 2);
    unsigned short* wbf   = (unsigned short*)carve((size_t)2048 * 1536 * 2);
    float*          dinv  = (float*)carve((size_t)NN * 4);
    float*          counts= (float*)carve((size_t)GG * 4);
    float*          bsum  = (float*)carve(2048 * 4);
    float*          bsq   = (float*)carve(2048 * 4);
    float*          scale = (float*)carve(2048 * 4);
    float*          shift = (float*)carve(2048 * 4);
    float*          z     = (float*)carve((size_t)GG * RDOUT * 4);
    unsigned short* zbf   = (unsigned short*)carve((size_t)GG * RDOUT * 2);
    float*          z1    = (float*)carve((size_t)GG * 1024 * 4);
    unsigned short* z1bf  = (unsigned short*)carve((size_t)GG * 1024 * 2);

    hipMemsetAsync(dinv,   0, (size_t)NN * 4,          stream);
    hipMemsetAsync(counts, 0, (size_t)GG * 4,          stream);
    hipMemsetAsync(z,      0, (size_t)GG * RDOUT * 4,  stream);

    deg_kernel   <<<cdiv(EE, 256), 256, 0, stream>>>(ei, dinv);
    counts_kernel<<<cdiv(NN, 256), 256, 0, stream>>>(batch, counts);
    dinv_kernel  <<<cdiv(NN, 256), 256, 0, stream>>>(dinv);

    // x -> bf16 (padded rows above NN hold don't-care bits; consumers guarded)
    tobf16_kernel<<<cdiv((long long)NN * 128 / 4, 256), 256, 0, stream>>>(
        x, xbf, (long long)NN * 128);

    const unsigned short* curbf = xbf;
    int din = 128, off = 0;
    for (int L = 0; L < 5; ++L) {
        int dout = (L < 4) ? 256 : 512;
        unsigned short* nextbf = hbf[L & 1];

        tobf16_kernel<<<cdiv((long long)din * dout / 4, 256), 256, 0, stream>>>(
            convW[L], wbf, (long long)din * dout);

        dim3 gg(NPAD / BM, dout / BN);
        gemm_bf16_wmma<<<gg, 256, 0, stream>>>(curbf, wbf, nullptr, lin, din, dout);

        hipMemsetAsync(agg,  0, (size_t)NN * dout * 4, stream);
        hipMemsetAsync(bsum, 0, (size_t)dout * 4,      stream);
        hipMemsetAsync(bsq,  0, (size_t)dout * 4,      stream);

        int lc = (dout == 256) ? 6 : 7;           // log2(dout/4)
        long long total = (long long)EE << lc;
        scatter_kernel<<<cdiv(total, 256), 256, 0, stream>>>(lin, ei, dinv, agg,
                                                             dout, lc, total);
        combine_stats_kernel<<<512, dout, 0, stream>>>(agg, lin, dinv, convB[L],
                                                       NN, dout, bsum, bsq);
        finalize_bn_kernel<<<cdiv(dout, 256), 256, 0, stream>>>(
            bsum, bsq, bnG[L], bnB[L], scale, shift, (float)NN, dout);
        bn_relu_pool_kernel<<<512, dout, 0, stream>>>(agg, scale, shift, batch,
                                                      nextbf, z + off, NN, dout);
        curbf = nextbf; din = dout; off += dout;
    }

    pool_div_kernel<<<cdiv(GG * RDOUT, 256), 256, 0, stream>>>(z, counts);

    // classifier: z[G,1536] @ w1 -> z1[G,1024], BN, relu, @ w2 -> out[G,2048]
    tobf16_kernel<<<cdiv((long long)GG * RDOUT / 4, 256), 256, 0, stream>>>(
        z, zbf, (long long)GG * RDOUT);
    tobf16_kernel<<<cdiv((long long)RDOUT * 1024 / 4, 256), 256, 0, stream>>>(
        w1, wbf, (long long)RDOUT * 1024);
    gemm_bf16_wmma<<<dim3(GG / BM, 1024 / BN), 256, 0, stream>>>(zbf, wbf, b1, z1,
                                                                 RDOUT, 1024);
    hipMemsetAsync(bsum, 0, 1024 * 4, stream);
    hipMemsetAsync(bsq,  0, 1024 * 4, stream);
    stats_kernel<<<256, 1024, 0, stream>>>(z1, GG, 1024, bsum, bsq);
    finalize_bn_kernel<<<4, 256, 0, stream>>>(bsum, bsq, g1, be1, scale, shift,
                                              (float)GG, 1024);
    bn_relu_kernel<<<256, 1024, 0, stream>>>(z1, scale, shift, GG, 1024);

    tobf16_kernel<<<cdiv((long long)GG * 1024 / 4, 256), 256, 0, stream>>>(
        z1, z1bf, (long long)GG * 1024);
    tobf16_kernel<<<cdiv((long long)1024 * 2048 / 4, 256), 256, 0, stream>>>(
        w2, wbf, (long long)1024 * 2048);
    gemm_bf16_wmma<<<dim3(GG / BM, 2048 / BN), 256, 0, stream>>>(z1bf, wbf, b2, out,
                                                                 1024, 2048);
}
